// SMOTERGAT_82497731822012
// MI455X (gfx1250) — matrix operand order repaired and checked by reading, not compiled, not run on hardware
//
#include <hip/hip_runtime.h>

// ---------------------------------------------------------------------------
// CDNA5 GAT pipeline: bf16 WMMA GEMMs (wave = 16x128 tile, 8 WMMA/k-step,
// software-pipelined double-buffered fragments) + CSR softmax aggregation.
// ---------------------------------------------------------------------------

typedef __attribute__((ext_vector_type(16))) __bf16 v16bf;
typedef __attribute__((ext_vector_type(2)))  __bf16 v2bf;
typedef __attribute__((ext_vector_type(8)))  float  v8f;

#define SCAN_BS 256
#define NEG_SLOPE 0.2f
#define AGG_TILE 64

__device__ __forceinline__ float lrelu(float v) {
  return v > 0.f ? v : NEG_SLOPE * v;
}

// pack two f32 into one dword of bf16 (RNE) -> v_cvt_pk_bf16_f32
__device__ __forceinline__ unsigned pk_bf16(float lo, float hi) {
  union { v2bf v; unsigned u; } r;
  r.v[0] = (__bf16)lo;
  r.v[1] = (__bf16)hi;
  return r.u;
}

__device__ __forceinline__ unsigned short f2bf_u(float f) {
  union { __bf16 b; unsigned short u; } r;
  r.b = (__bf16)f;
  return r.u;
}

union V16BF { v16bf v; unsigned d[8]; };

// ---------------------------------------------------------------------------
// CSR construction (reused by all 4 GAT stages)
// ---------------------------------------------------------------------------
__global__ void k_init_deg_fill(int* __restrict__ deg, int* __restrict__ fill, int N) {
  int i = blockIdx.x * blockDim.x + threadIdx.x;
  if (i < N) { deg[i] = 1; fill[i] = 0; }   // deg starts at 1: self loop
}

__global__ void k_count_deg(const int* __restrict__ eidx, int* __restrict__ deg, int E) {
  int t = blockIdx.x * blockDim.x + threadIdx.x;
  if (t < E) atomicAdd(&deg[eidx[E + t]], 1);
}

__global__ void k_scan_block(const int* __restrict__ in, int* __restrict__ incl,
                             int* __restrict__ bsum, int n) {
  __shared__ int s[SCAN_BS];
  int i = blockIdx.x * SCAN_BS + threadIdx.x;
  s[threadIdx.x] = (i < n) ? in[i] : 0;
  __syncthreads();
  for (int off = 1; off < SCAN_BS; off <<= 1) {
    int t = (threadIdx.x >= off) ? s[threadIdx.x - off] : 0;
    __syncthreads();
    s[threadIdx.x] += t;
    __syncthreads();
  }
  if (i < n) incl[i] = s[threadIdx.x];
  if (threadIdx.x == SCAN_BS - 1) bsum[blockIdx.x] = s[threadIdx.x];
}

__global__ void k_scan_sums(int* __restrict__ bsum, int nb) {  // nb <= 256
  __shared__ int s[SCAN_BS];
  s[threadIdx.x] = (threadIdx.x < nb) ? bsum[threadIdx.x] : 0;
  __syncthreads();
  for (int off = 1; off < SCAN_BS; off <<= 1) {
    int t = (threadIdx.x >= off) ? s[threadIdx.x - off] : 0;
    __syncthreads();
    s[threadIdx.x] += t;
    __syncthreads();
  }
  if (threadIdx.x < nb) bsum[threadIdx.x] = s[threadIdx.x];
}

__global__ void k_scan_finalize(const int* __restrict__ incl, const int* __restrict__ deg,
                                const int* __restrict__ bscan, int* __restrict__ rowptr,
                                int n, int nb) {
  int i = blockIdx.x * blockDim.x + threadIdx.x;
  if (i < n) {
    int blk = i / SCAN_BS;
    int off = blk ? bscan[blk - 1] : 0;
    rowptr[i] = incl[i] - deg[i] + off;      // exclusive scan
  }
  if (i == n) rowptr[n] = bscan[nb - 1];
}

__global__ void k_scatter(const int* __restrict__ eidx, const int* __restrict__ rowptr,
                          int* __restrict__ fill, int* __restrict__ csr, int E, int N) {
  int t = blockIdx.x * blockDim.x + threadIdx.x;
  if (t >= E + N) return;
  int s, d;
  if (t < E) { s = eidx[t]; d = eidx[E + t]; }
  else       { s = d = t - E; }              // self loop
  int pos = rowptr[d] + atomicAdd(&fill[d], 1);
  csr[pos] = s;
}

// ---------------------------------------------------------------------------
// Weight packing: W[K,128] f32 -> bf16 in the B-operand lane order used here (contiguous k, same as A).
// B (32x16 per tile): lane L holds column n=L&15; element e -> K = 16*(L>>4)+e.
// ---------------------------------------------------------------------------
__global__ void k_pack_w(const float* __restrict__ W, unsigned short* __restrict__ Wp, int K) {
  int t = blockIdx.x * blockDim.x + threadIdx.x;
  if (t >= K * 128) return;
  int e    = t & 15;
  int r    = t >> 4;
  int lane = r & 31; r >>= 5;
  int nt   = r & 7;
  int kt   = r >> 3;
  int n = (nt << 4) + (lane & 15);
  int k = (kt << 5) + ((lane >> 4) << 4) + e;
  Wp[t] = f2bf_u(W[k * 128 + n]);
}

// ---------------------------------------------------------------------------
// GEMM: C[M,128] = A[M,K(lda)] x W.   bf16 WMMA, f32 accumulate.
// 256 threads = 8 waves; wave w owns rows [128*blk + 16w, +16) x all 128 cols.
// Software pipeline: fragment sets for step k+1 are loaded while the 8 WMMAs
// of step k execute (KT is even: 48 or 4; last two steps peeled).
// remap!=0: feature-view-2 column map (k<768 -> k, else k+768).
// ---------------------------------------------------------------------------
__device__ __forceinline__ v16bf load_afrag(const float* __restrict__ ap, int c0, int c1) {
  float4 f0 = *(const float4*)(ap + c0);
  float4 f1 = *(const float4*)(ap + c0 + 4);
  float4 f2 = *(const float4*)(ap + c1);
  float4 f3 = *(const float4*)(ap + c1 + 4);
  V16BF a;
  a.d[0] = pk_bf16(f0.x, f0.y); a.d[1] = pk_bf16(f0.z, f0.w);
  a.d[2] = pk_bf16(f1.x, f1.y); a.d[3] = pk_bf16(f1.z, f1.w);
  a.d[4] = pk_bf16(f2.x, f2.y); a.d[5] = pk_bf16(f2.z, f2.w);
  a.d[6] = pk_bf16(f3.x, f3.y); a.d[7] = pk_bf16(f3.z, f3.w);
  return a.v;
}

struct Frag {
  v16bf a;
  v16bf b[8];
};

__device__ __forceinline__ Frag load_step(const float* __restrict__ arow,
                                          const unsigned short* __restrict__ bb,
                                          int kt, int kg, int remap) {
  Frag f;
  int kb0 = (kt << 5) + (kg << 4);
  int kb1 = kb0 + 8;
  int c0 = (remap && kb0 >= 768) ? kb0 + 768 : kb0;
  int c1 = (remap && kb1 >= 768) ? kb1 + 768 : kb1;
  f.a = load_afrag(arow, c0, c1);
  const unsigned short* bk = bb + (size_t)kt * 4096;
  #pragma unroll
  for (int j = 0; j < 8; ++j) f.b[j] = *(const v16bf*)(bk + (size_t)j * 512);
  return f;
}

__device__ __forceinline__ void do_wmma8(v8f acc[8], const Frag& f) {
  #pragma unroll
  for (int j = 0; j < 8; ++j)
    acc[j] = __builtin_amdgcn_wmma_f32_16x16x32_bf16(
        false, f.a, false, f.b[j], (short)0, acc[j], false, false);
}

__global__ void __launch_bounds__(256) k_gemm_wmma(
    const float* __restrict__ A, const unsigned short* __restrict__ Bp,
    float* __restrict__ C, int M, int K, int lda, int remap) {
  const int lane = threadIdx.x & 31;
  const int wv   = threadIdx.x >> 5;           // 0..7 -> row subtile
  const int m    = lane & 15;
  const int kg   = lane >> 4;
  const int rowbase = blockIdx.x * 128 + wv * 16;   // uniform per wave
  const int KT = K >> 5;                       // even (48 or 4)

  int r = rowbase + m;
  if (r >= M) r = M - 1;                       // data clamp; EXEC stays all-ones
  const float* arow = A + (size_t)r * lda;
  const unsigned short* bb = Bp + (size_t)lane * 16;

  v8f acc[8] = {};

  Frag f0 = load_step(arow, bb, 0, kg, remap);
  int kt = 0;
  for (; kt + 2 < KT; kt += 2) {
    Frag f1 = load_step(arow, bb, kt + 1, kg, remap);
    do_wmma8(acc, f0);
    f0 = load_step(arow, bb, kt + 2, kg, remap);
    do_wmma8(acc, f1);
  }
  {                                            // tail: steps KT-2, KT-1
    Frag f1 = load_step(arow, bb, kt + 1, kg, remap);
    do_wmma8(acc, f0);
    do_wmma8(acc, f1);
  }

  // C/D layout: VGPR r holds element (M = 8*(lane>>4)+r, N = lane&15)
  const int nn = lane & 15;
  const int mh = lane >> 4;
  if (rowbase + 16 <= M) {                     // wave-uniform fast path
    float* cp0 = C + (size_t)(rowbase + (mh << 3)) * 128 + nn;
    #pragma unroll
    for (int j = 0; j < 8; ++j) {
      float* cp = cp0 + (j << 4);
      #pragma unroll
      for (int rr = 0; rr < 8; ++rr) cp[(size_t)rr * 128] = acc[j][rr];
    }
  } else {
    #pragma unroll
    for (int j = 0; j < 8; ++j) {
      #pragma unroll
      for (int rr = 0; rr < 8; ++rr) {
        int row = rowbase + (mh << 3) + rr;
        if (row < M) C[(size_t)row * 128 + (j << 4) + nn] = acc[j][rr];
      }
    }
  }
}

// ---------------------------------------------------------------------------
// Attention coefficients:  asrc[n,h] = sum_c h[n,h,c]*a_src[h,c]  (adst alike)
// ---------------------------------------------------------------------------
__global__ void k_attn_dots(const float* __restrict__ h,
                            const float* __restrict__ a_src, const float* __restrict__ a_dst,
                            float* __restrict__ asrc, float* __restrict__ adst,
                            int N, int H, int C) {
  int t = blockIdx.x * blockDim.x + threadIdx.x;
  if (t >= N * H) return;
  int hh = t % H;
  const float* hp = h + (size_t)(t / H) * 128 + hh * C;
  float s1 = 0.f, s2 = 0.f;
  for (int c = 0; c < C; ++c) {
    float v = hp[c];
    s1 += v * a_src[hh * C + c];
    s2 += v * a_dst[hh * C + c];
  }
  asrc[t] = s1; adst[t] = s2;
}

// segment max + softmax denominator per (dst node, head)
__global__ void k_seg_stats(const int* __restrict__ rowptr, const int* __restrict__ csr,
                            const float* __restrict__ asrc, const float* __restrict__ adst,
                            float* __restrict__ emax, float* __restrict__ denom,
                            int N, int H) {
  int t = blockIdx.x * blockDim.x + threadIdx.x;
  if (t >= N * H) return;
  int n = t / H, hh = t % H;
  float ad = adst[t];
  int jb = rowptr[n], je = rowptr[n + 1];
  float mx = -1e30f;
  for (int j = jb; j < je; ++j)
    mx = fmaxf(mx, lrelu(asrc[csr[j] * H + hh] + ad));
  float s = 0.f;
  for (int j = jb; j < je; ++j)
    s += __expf(lrelu(asrc[csr[j] * H + hh] + ad) - mx);
  emax[t] = mx; denom[t] = s;
}

// out[n,c] = (sum_{s in N(n)} w_sn * h[s,c]) / denom[n,h(c)] + bias[c]
// one block per node, 128 threads (one per channel). Edge list + per-(edge,head)
// softmax weights staged through LDS; inner loop = coalesced 512B gathers of
// h[src] + LDS reads.
__global__ void __launch_bounds__(128) k_aggregate(
    const int* __restrict__ rowptr, const int* __restrict__ csr,
    const float* __restrict__ h,
    const float* __restrict__ asrc, const float* __restrict__ adst,
    const float* __restrict__ emax, const float* __restrict__ denom,
    const float* __restrict__ bias, float* __restrict__ out,
    int H, int C, int accum) {
  __shared__ int   s_src[AGG_TILE];
  __shared__ float s_w[AGG_TILE * 4];          // H <= 4
  const int n = blockIdx.x;
  const int c = threadIdx.x;                   // 0..127
  const int hh = c / C;
  const float dn = denom[n * H + hh];
  const int jb = rowptr[n], je = rowptr[n + 1];
  float acc = 0.f;
  for (int base = jb; base < je; base += AGG_TILE) {
    int cnt = min(AGG_TILE, je - base);
    __syncthreads();                           // protect LDS reuse
    if (c < cnt) s_src[c] = csr[base + c];
    __syncthreads();
    for (int u = c; u < cnt * H; u += 128) {   // one exp per (edge, head)
      int e = u / H, h2 = u - e * H;
      int t2 = n * H + h2;
      float v = lrelu(asrc[s_src[e] * H + h2] + adst[t2]);
      s_w[u] = __expf(v - emax[t2]);
    }
    __syncthreads();
    for (int e = 0; e < cnt; ++e)
      acc += s_w[e * H + hh] * h[(size_t)s_src[e] * 128 + c];
  }
  float val = acc / dn + bias[c];
  size_t o = (size_t)n * 128 + c;
  if (accum) out[o] += val; else out[o] = val;
}

// ---------------------------------------------------------------------------
extern "C" void kernel_launch(void* const* d_in, const int* in_sizes, int n_in,
                              void* d_out, int out_size, void* d_ws, size_t ws_size,
                              hipStream_t stream) {
  const float* feature = (const float*)d_in[0];
  const int*   eidx    = (const int*)d_in[1];
  const float* W1  = (const float*)d_in[3];
  const float* as1 = (const float*)d_in[4];
  const float* ad1 = (const float*)d_in[5];
  const float* b1  = (const float*)d_in[6];
  const float* W2  = (const float*)d_in[7];
  const float* as2 = (const float*)d_in[8];
  const float* ad2 = (const float*)d_in[9];
  const float* b2  = (const float*)d_in[10];
  const float* W12  = (const float*)d_in[11];
  const float* as12 = (const float*)d_in[12];
  const float* ad12 = (const float*)d_in[13];
  const float* b12  = (const float*)d_in[14];
  const float* W22  = (const float*)d_in[15];
  const float* as22 = (const float*)d_in[16];
  const float* ad22 = (const float*)d_in[17];
  const float* b22  = (const float*)d_in[18];

  const int N = in_sizes[0] / 2304;
  const int E = in_sizes[1] / 2;

  // ---- workspace carve (256B aligned) ----
  char* wp = (char*)d_ws;
  auto carve = [&](size_t bytes) -> void* {
    void* q = (void*)wp; wp += (bytes + 255) & ~(size_t)255; return q;
  };
  float* h1 = (float*)carve((size_t)N * 128 * 4);
  float* h2 = (float*)carve((size_t)N * 128 * 4);
  float* x1 = (float*)carve((size_t)N * 128 * 4);
  float* x2 = (float*)carve((size_t)N * 128 * 4);
  unsigned short* W1p  = (unsigned short*)carve(1536 * 128 * 2);
  unsigned short* W2p  = (unsigned short*)carve(1536 * 128 * 2);
  unsigned short* W12p = (unsigned short*)carve(128 * 128 * 2);
  unsigned short* W22p = (unsigned short*)carve(128 * 128 * 2);
  float* asrc  = (float*)carve((size_t)N * 4 * 4);
  float* adst  = (float*)carve((size_t)N * 4 * 4);
  float* emax  = (float*)carve((size_t)N * 4 * 4);
  float* denom = (float*)carve((size_t)N * 4 * 4);
  int* deg    = (int*)carve((size_t)N * 4);
  int* fill   = (int*)carve((size_t)N * 4);
  int* incl   = (int*)carve((size_t)N * 4);
  int* bsum   = (int*)carve(SCAN_BS * 4);
  int* rowptr = (int*)carve((size_t)(N + 1) * 4);
  int* csr    = (int*)carve((size_t)(E + N) * 4);

  const int TB = 256;
  auto cdiv = [](int a, int b) { return (a + b - 1) / b; };

  // ---- CSR build (once, reused by all 4 GAT stages) ----
  k_init_deg_fill<<<cdiv(N, TB), TB, 0, stream>>>(deg, fill, N);
  k_count_deg<<<cdiv(E, TB), TB, 0, stream>>>(eidx, deg, E);
  int nb = cdiv(N, SCAN_BS);
  k_scan_block<<<nb, SCAN_BS, 0, stream>>>(deg, incl, bsum, N);
  k_scan_sums<<<1, SCAN_BS, 0, stream>>>(bsum, nb);
  k_scan_finalize<<<cdiv(N + 1, TB), TB, 0, stream>>>(incl, deg, bsum, rowptr, N, nb);
  k_scatter<<<cdiv(E + N, TB), TB, 0, stream>>>(eidx, rowptr, fill, csr, E, N);

  // ---- weight packing ----
  k_pack_w<<<cdiv(1536 * 128, TB), TB, 0, stream>>>(W1, W1p, 1536);
  k_pack_w<<<cdiv(1536 * 128, TB), TB, 0, stream>>>(W2, W2p, 1536);
  k_pack_w<<<cdiv(128 * 128, TB), TB, 0, stream>>>(W12, W12p, 128);
  k_pack_w<<<cdiv(128 * 128, TB), TB, 0, stream>>>(W22, W22p, 128);

  const int gm = cdiv(N, 128);

  // ---- layer 1 GEMMs (feature views -> h1/h2) ----
  k_gemm_wmma<<<gm, 256, 0, stream>>>(feature, W1p, h1, N, 1536, 2304, 0);
  k_gemm_wmma<<<gm, 256, 0, stream>>>(feature, W2p, h2, N, 1536, 2304, 1);

  // ---- GAT stage 1 (H=4, C=32) ----
  k_attn_dots<<<cdiv(N * 4, TB), TB, 0, stream>>>(h1, as1, ad1, asrc, adst, N, 4, 32);
  k_seg_stats<<<cdiv(N * 4, TB), TB, 0, stream>>>(rowptr, csr, asrc, adst, emax, denom, N, 4);
  k_aggregate<<<N, 128, 0, stream>>>(rowptr, csr, h1, asrc, adst, emax, denom, b1, x1, 4, 32, 0);

  k_attn_dots<<<cdiv(N * 4, TB), TB, 0, stream>>>(h2, as2, ad2, asrc, adst, N, 4, 32);
  k_seg_stats<<<cdiv(N * 4, TB), TB, 0, stream>>>(rowptr, csr, asrc, adst, emax, denom, N, 4);
  k_aggregate<<<N, 128, 0, stream>>>(rowptr, csr, h2, asrc, adst, emax, denom, b2, x2, 4, 32, 0);

  // ---- layer 2 (H=1, C=128); reuse h1/h2 as scratch for projected features ----
  k_gemm_wmma<<<gm, 256, 0, stream>>>(x1, W12p, h1, N, 128, 128, 0);
  k_attn_dots<<<cdiv(N, TB), TB, 0, stream>>>(h1, as12, ad12, asrc, adst, N, 1, 128);
  k_seg_stats<<<cdiv(N, TB), TB, 0, stream>>>(rowptr, csr, asrc, adst, emax, denom, N, 1);
  k_aggregate<<<N, 128, 0, stream>>>(rowptr, csr, h1, asrc, adst, emax, denom, b12,
                                     (float*)d_out, 1, 128, 0);

  k_gemm_wmma<<<gm, 256, 0, stream>>>(x2, W22p, h2, N, 128, 128, 0);
  k_attn_dots<<<cdiv(N, TB), TB, 0, stream>>>(h2, as22, ad22, asrc, adst, N, 1, 128);
  k_seg_stats<<<cdiv(N, TB), TB, 0, stream>>>(rowptr, csr, asrc, adst, emax, denom, N, 1);
  k_aggregate<<<N, 128, 0, stream>>>(rowptr, csr, h2, asrc, adst, emax, denom, b22,
                                     (float*)d_out, 1, 128, 1);
}
